// MultiHeadAttention_35794257444885
// MI455X (gfx1250) — matrix operand order repaired and checked by reading, not compiled, not run on hardware
//
#include <hip/hip_runtime.h>
#include <hip/hip_bf16.h>

typedef __attribute__((ext_vector_type(16))) _Float16 v16h;
typedef __attribute__((ext_vector_type(8)))  float    v8f;

#define WMMA_F16(a,b,c) __builtin_amdgcn_wmma_f32_16x16x32_f16( \
    false, (a), false, (b), (short)0, (c), false, false)

constexpr int Bn  = 1024;   // batch (incl. windows)
constexpr int Sn  = 64;     // window seq len
constexpr int Dn  = 768;    // hidden
constexpr int Hn  = 12;     // heads
constexpr int NWn = 64;     // num windows
constexpr int HDn = Dn / Hn;      // 64
constexpr int BSn = Bn * Sn;      // 65536

// --- CDNA5 async load-to-LDS (guarded; falls back to plain LDS stores) ---
#if defined(__AMDGCN__) && __has_builtin(__builtin_amdgcn_global_load_async_to_lds_b128)
#define HAVE_ASYNC_LDS 1
typedef int v4i_gnu __attribute__((vector_size(16)));           // matches builtin param
typedef __attribute__((address_space(3))) v4i_gnu as3_v4i;
#define ASYNC_COPY16(gp, lp) \
  __builtin_amdgcn_global_load_async_to_lds_b128( \
      (v4i_gnu*)(unsigned long long)(gp), (as3_v4i*)(lp), 0, 0)
__device__ __forceinline__ void wait_async0() {
#if __has_builtin(__builtin_amdgcn_s_wait_asynccnt)
  __builtin_amdgcn_s_wait_asynccnt(0);
#else
  asm volatile("s_wait_asynccnt 0x0" ::: "memory");
#endif
}
#else
#define HAVE_ASYNC_LDS 0
#endif

// ---- WMMA fragment loaders (contiguous k order, same for A and B, not the ISA table order; wave32) ----
// A 16-bit 16x32: lane = M|khalf<<4 ; halves 0..15 -> K = khalf*16+h.
__device__ __forceinline__ v16h ldA(const _Float16* p) {
  union { v16h v; uint4 q[2]; } u;
  u.q[0] = *(const uint4*)(p);
  u.q[1] = *(const uint4*)(p + 8);
  return u.v;
}
// B 16-bit 32x16: lane = N|khalf<<4 ; halves 0..15 -> K = khalf*16+h
// (16 contiguous halves when column N is contiguous over K in LDS).
__device__ __forceinline__ v16h ldB(const _Float16* p) {
  union { v16h v; uint4 q[2]; } u;
  u.q[0] = *(const uint4*)(p);
  u.q[1] = *(const uint4*)(p + 8);
  return u.v;
}

// ---------------------------------------------------------------------
// Tiled GEMM: Y[M=BS x N=D] = X[BS x D] @ W[D x D]
// 64x64 tile per block, 128 threads (4 waves), double-buffered LDS.
// Grid: x = column tile (12, fast-varying) so consecutive blocks share
// the same 196KB X row-strip via L2; W (2.4MB) is L2-resident.
// ---------------------------------------------------------------------
template<bool IN_F16, bool OUT_SPLIT>
__global__ __launch_bounds__(128, 2)
void gemm64(const void* __restrict__ Xv, const float* __restrict__ W,
            void* __restrict__ Yv)
{
  __shared__ __align__(16) _Float16 At [2][64 * 40];  // rows M, K 0..31, pad->40
  __shared__ __align__(16) _Float16 BtT[2][64 * 40];  // rows N, K 0..31, pad->40

  const int tid   = threadIdx.x;
  const int lane  = tid & 31;
  const int wave  = tid >> 5;
  const int m15   = lane & 15;
  const int khalf = lane >> 4;
  const int colBase = blockIdx.x * 64;   // x = column tile (12)
  const int rowBase = blockIdx.y * 64;   // y = row tile (1024)

  const float*    Xf = (const float*)Xv;
  const _Float16* Xh = (const _Float16*)Xv;

  // --- staging helpers (vectorized global loads) ---
  auto stage = [&](int kt, int buf) {
    const int k0 = kt * 32;
    if (IN_F16) {
      // A tile 64x32 halves: 256 x 16B chunks
      for (int i = tid; i < 256; i += 128) {
        int m = i >> 2, c16 = i & 3;             // 4 x 16B chunks per row
        *(uint4*)&At[buf][m * 40 + c16 * 8] =
            *(const uint4*)&Xh[(size_t)(rowBase + m) * Dn + k0 + c16 * 8];
      }
    } else {
      // A tile 64x32 fp32 -> f16: 512 float4 chunks
      for (int i = tid; i < 512; i += 128) {
        int m = i >> 3, c4 = i & 7;              // 8 x float4 per row
        float4 v = *(const float4*)&Xf[(size_t)(rowBase + m) * Dn + k0 + c4 * 4];
        union { _Float16 h[4]; uint2 u2; } u;
        u.h[0] = (_Float16)v.x; u.h[1] = (_Float16)v.y;
        u.h[2] = (_Float16)v.z; u.h[3] = (_Float16)v.w;
        *(uint2*)&At[buf][m * 40 + c4 * 4] = u.u2;
      }
    }
    // B tile 32x64 fp32 -> f16, stored transposed BtT[n][k]
    for (int i = tid; i < 512; i += 128) {
      int r = i >> 4, c4 = i & 15;               // 16 x float4 per W row
      float4 v = *(const float4*)&W[(size_t)(k0 + r) * Dn + colBase + c4 * 4];
      BtT[buf][(c4 * 4 + 0) * 40 + r] = (_Float16)v.x;
      BtT[buf][(c4 * 4 + 1) * 40 + r] = (_Float16)v.y;
      BtT[buf][(c4 * 4 + 2) * 40 + r] = (_Float16)v.z;
      BtT[buf][(c4 * 4 + 3) * 40 + r] = (_Float16)v.w;
    }
  };

  v8f acc[4] = {};
  stage(0, 0);

  constexpr int NKT = Dn / 32;  // 24
  for (int kt = 0; kt < NKT; ++kt) {
    const int cur = kt & 1;
    __syncthreads();                       // buf[cur] ready; buf[cur^1] free
    if (kt + 1 < NKT) stage(kt + 1, cur ^ 1);

    v16h a = ldA(&At[cur][(wave * 16 + m15) * 40 + khalf * 16]);
#pragma unroll
    for (int nt = 0; nt < 4; ++nt) {
      v16h b = ldB(&BtT[cur][(nt * 16 + m15) * 40 + khalf * 16]);
      acc[nt] = WMMA_F16(a, b, acc[nt]);
    }
  }

  // C/D layout: element r -> M = 8*khalf + r, N = m15
  const int mrow = rowBase + wave * 16 + khalf * 8;
#pragma unroll
  for (int nt = 0; nt < 4; ++nt) {
    const int ncol = colBase + nt * 16 + m15;
#pragma unroll
    for (int r = 0; r < 8; ++r) {
      const int M = mrow + r;
      const float v = acc[nt][r];
      if (OUT_SPLIT) {
        const int b = M >> 6, s = M & 63;        // S = 64
        const int h = ncol >> 6, d = ncol & 63;  // HD = 64
        ((_Float16*)Yv)[(((size_t)(b * Hn + h)) * Sn + s) * HDn + d] = (_Float16)v;
      } else {
        ((float*)Yv)[(size_t)M * Dn + ncol] = v;
      }
    }
  }
}

// ---------------------------------------------------------------------
// Attention core: one block per (b,h). Q,K,V tiles are 64x64 f16.
// scores = QK^T/8 + window_mask, valid_len mask, softmax, O = P@V.
// Q/K staged via CDNA5 async load-to-LDS when available.
// ---------------------------------------------------------------------
__global__ __launch_bounds__(128, 2)
void attn64(const _Float16* __restrict__ Q, const _Float16* __restrict__ K,
            const _Float16* __restrict__ V, const float* __restrict__ wmask,
            const int* __restrict__ vlens, _Float16* __restrict__ Oa)
{
  __shared__ __align__(16) _Float16 Qs [64 * 72];  // Q tile; reused as P
  __shared__ __align__(16) _Float16 Ks [64 * 72];  // K tile (row k, col d)
  __shared__ __align__(16) _Float16 VsT[64 * 72];  // V transposed (row d, col k)
  __shared__ __align__(16) float    Ssc[64 * 64];  // scores f32

  const int tid   = threadIdx.x;
  const int lane  = tid & 31;
  const int wave  = tid >> 5;
  const int m15   = lane & 15;
  const int khalf = lane >> 4;

  const int bh = blockIdx.x;
  const int b  = bh / Hn;
  const int h  = bh % Hn;
  const size_t base = (size_t)bh * (Sn * HDn);
  const int   vl = vlens[b];
  const float* wm = wmask + (size_t)(b % NWn) * (Sn * Sn);

#if HAVE_ASYNC_LDS
  // Q/K rows are contiguous 128B; copy as 16B chunks, row stride padded to 144B.
  for (int i = tid; i < 512; i += 128) {
    int s = i >> 3, c = i & 7;                    // 8 x 16B chunks per row
    ASYNC_COPY16(Q + base + s * 64 + c * 8, &Qs[s * 72 + c * 8]);
    ASYNC_COPY16(K + base + s * 64 + c * 8, &Ks[s * 72 + c * 8]);
  }
#else
  {
    const unsigned int* Qg = (const unsigned int*)(Q + base);
    const unsigned int* Kg = (const unsigned int*)(K + base);
    for (int i = tid; i < 64 * 32; i += 128) {
      int s = i >> 5, c2 = i & 31;
      *(unsigned int*)&Qs[s * 72 + c2 * 2] = Qg[i];
      *(unsigned int*)&Ks[s * 72 + c2 * 2] = Kg[i];
    }
  }
#endif
  // V transposed on stage-in: VsT[d][k]  (dword global loads, b16 LDS scatter)
  {
    const unsigned int* Vg = (const unsigned int*)(V + base);
    for (int i = tid; i < 2048; i += 128) {
      int k = i >> 5, d2 = i & 31;
      union { unsigned int u; _Float16 hh[2]; } u;
      u.u = Vg[i];
      VsT[(d2 * 2 + 0) * 72 + k] = u.hh[0];
      VsT[(d2 * 2 + 1) * 72 + k] = u.hh[1];
    }
  }
#if HAVE_ASYNC_LDS
  wait_async0();
#endif
  __syncthreads();

  const int m0 = wave * 16;               // this wave's query-row strip

  // ---- scores = Q @ K^T : B[d][k] = K[k][d] -> column k contiguous in Ks row
  {
    v8f acc[4] = {};
#pragma unroll
    for (int ks = 0; ks < 2; ++ks) {
      v16h a = ldA(&Qs[(m0 + m15) * 72 + ks * 32 + khalf * 16]);
#pragma unroll
      for (int nt = 0; nt < 4; ++nt) {
        v16h bf = ldB(&Ks[(nt * 16 + m15) * 72 + ks * 32 + khalf * 16]);
        acc[nt] = WMMA_F16(a, bf, acc[nt]);
      }
    }
#pragma unroll
    for (int nt = 0; nt < 4; ++nt) {
      const int kc = nt * 16 + m15;
#pragma unroll
      for (int r = 0; r < 8; ++r) {
        const int qr = m0 + khalf * 8 + r;
        float sv = acc[nt][r] * 0.125f + wm[qr * 64 + kc];  // 1/sqrt(64)
        if (kc >= vl) sv = -1.0e6f;
        Ssc[qr * 64 + kc] = sv;
      }
    }
  }
  __syncthreads();

  // ---- softmax: one row per thread, write P (f16) into Qs storage
  if (tid < 64) {
    float* row = &Ssc[tid * 64];
    float mx = row[0];
    for (int j = 1; j < 64; ++j) mx = fmaxf(mx, row[j]);
    float sum = 0.0f;
    for (int j = 0; j < 64; ++j) { float e = __expf(row[j] - mx); row[j] = e; sum += e; }
    const float inv = 1.0f / sum;
    for (int j = 0; j < 64; ++j) Qs[tid * 72 + j] = (_Float16)(row[j] * inv);
  }
  __syncthreads();

  // ---- O = P @ V : B[k][d] = V[k][d] -> column d contiguous in VsT row
  {
    v8f acc[4] = {};
#pragma unroll
    for (int ks = 0; ks < 2; ++ks) {
      v16h a = ldA(&Qs[(m0 + m15) * 72 + ks * 32 + khalf * 16]);
#pragma unroll
      for (int nt = 0; nt < 4; ++nt) {
        v16h bf = ldB(&VsT[(nt * 16 + m15) * 72 + ks * 32 + khalf * 16]);
        acc[nt] = WMMA_F16(a, bf, acc[nt]);
      }
    }
#pragma unroll
    for (int nt = 0; nt < 4; ++nt) {
      const int d = nt * 16 + m15;
#pragma unroll
      for (int r = 0; r < 8; ++r) {
        const int s = m0 + khalf * 8 + r;
        Oa[((size_t)(b * Sn + s)) * Dn + h * HDn + d] = (_Float16)acc[nt][r];
      }
    }
  }
}

// ---------------------------------------------------------------------
extern "C" void kernel_launch(void* const* d_in, const int* in_sizes, int n_in,
                              void* d_out, int out_size, void* d_ws, size_t ws_size,
                              hipStream_t stream)
{
  (void)in_sizes; (void)n_in; (void)out_size; (void)ws_size;
  const void*  dq  = d_in[0];                     // queries  fp32 (B,S,D)
  const void*  dk  = d_in[1];                     // keys     fp32
  const void*  dv  = d_in[2];                     // values   fp32
  const float* Wq  = (const float*)d_in[3];
  const float* Wk  = (const float*)d_in[4];
  const float* Wv  = (const float*)d_in[5];
  const float* Wo  = (const float*)d_in[6];
  const float* wmask = (const float*)d_in[7];     // (NW,S,S)
  const int*   vlens = (const int*)d_in[8];       // (B,)

  const size_t plane = (size_t)BSn * Dn * sizeof(_Float16);  // ~100.7 MB
  _Float16* Qh = (_Float16*)d_ws;
  _Float16* Kh = (_Float16*)((char*)d_ws + 1 * plane);
  _Float16* Vh = (_Float16*)((char*)d_ws + 2 * plane);
  _Float16* Oh = (_Float16*)((char*)d_ws + 3 * plane);

  dim3 ggrid(Dn / 64, BSn / 64);  // x = 12 column tiles (fast), y = 1024 rows

  gemm64<false, true><<<ggrid, 128, 0, stream>>>(dq, Wq, Qh);
  gemm64<false, true><<<ggrid, 128, 0, stream>>>(dk, Wk, Kh);
  gemm64<false, true><<<ggrid, 128, 0, stream>>>(dv, Wv, Vh);

  attn64<<<Bn * Hn, 128, 0, stream>>>(Qh, Kh, Vh, wmask, vlens, Oh);

  gemm64<true, false><<<ggrid, 128, 0, stream>>>(Oh, Wo, d_out);
}
